// MultiHeadAttention_31490700215070
// MI455X (gfx1250) — matrix operand order repaired and checked by reading, not compiled, not run on hardware
//
#include <hip/hip_runtime.h>
#include <hip/hip_bf16.h>

// Problem constants (from reference): B=4, S=2048, D=768, H=12, HD=64
#define B_  4
#define S_  2048
#define D_  768
#define H_  12
#define HD_ 64

typedef __attribute__((ext_vector_type(16))) __bf16 v16bf;
typedef __attribute__((ext_vector_type(8)))  __bf16 v8bf;
typedef __attribute__((ext_vector_type(4)))  __bf16 v4bf;
typedef __attribute__((ext_vector_type(8)))  float  v8f;

__device__ __forceinline__ v16bf cat16(v8bf lo, v8bf hi) {
  return __builtin_shufflevector(lo, hi, 0,1,2,3,4,5,6,7,8,9,10,11,12,13,14,15);
}

__device__ __forceinline__ v8f wmma_bf16(v16bf a, v16bf b, v8f c) {
  // D = A(16x32 bf16) * B(32x16 bf16) + C(16x16 f32)
  return __builtin_amdgcn_wmma_f32_16x16x32_bf16(false, a, false, b, (short)0, c,
                                                 false, false);
}

// ---------------------------------------------------------------------------
// Kernel 0a: prepack a [768,768] fp32 weight into bf16 B-fragment order.
// For (k-chunk kc, n-tile nt): a 16x32 column-major tile so that lane lm's
// 16 K-elements (K = half*16 + e) are contiguous:
//   dst = ((kc*48 + nt)*16 + n%16)*32 + k%32
// ---------------------------------------------------------------------------
__global__ __launch_bounds__(256) void pack_w_kernel(const float* __restrict__ W,
                                                     __bf16* __restrict__ Wp) {
  const int t = blockIdx.x * 256 + threadIdx.x;        // 0 .. 768*768-1
  const int k = t / D_;
  const int n = t % D_;                                 // coalesced read along n
  const int dst = (((k >> 5) * (D_ / 16) + (n >> 4)) * 16 + (n & 15)) * 32 + (k & 31);
  Wp[dst] = (__bf16)W[t];
}

// Kernel 0b: convert x (fp32) to bf16, same row-major layout (A-side operand).
__global__ __launch_bounds__(256) void pack_x_kernel(const float* __restrict__ x,
                                                     __bf16* __restrict__ xb) {
  const size_t i = ((size_t)blockIdx.x * 256 + threadIdx.x) * 4;
  const float4 f = *(const float4*)(x + i);
  v4bf o = { (__bf16)f.x, (__bf16)f.y, (__bf16)f.z, (__bf16)f.w };
  *(v4bf*)(xb + i) = o;
}

// ---------------------------------------------------------------------------
// Kernel 1: fused QKV projection (all operands pre-packed bf16).
//   q = x@Wq + bq  -> qws  [B,H,S,HD] bf16
//   k = x@Wk       -> kws  [B,H,S,HD] bf16
//   v = x@Wv + bv  -> vtws [B,H,HD,S] bf16   (transposed for PV B-fragments)
// One wave computes a 16x64 output tile for all three matrices (A reuse x3).
// grid = (512/8, 12), block = 256 (8 waves).
// ---------------------------------------------------------------------------
__global__ __launch_bounds__(256) void proj_qkv_kernel(
    const __bf16* __restrict__ xb,
    const __bf16* __restrict__ Wqp, const float* __restrict__ bq,
    const __bf16* __restrict__ Wkp,
    const __bf16* __restrict__ Wvp, const float* __restrict__ bv,
    __bf16* __restrict__ qws, __bf16* __restrict__ kws, __bf16* __restrict__ vtws)
{
  const int lane = threadIdx.x & 31;
  const int wv   = threadIdx.x >> 5;
  const int half = lane >> 4;
  const int lm   = lane & 15;
  const int mt   = blockIdx.x * 8 + wv;     // M tile: 0..511 (rows of x)
  const int g    = blockIdx.y;              // head / 64-col group: 0..11
  const int arow = mt * 16 + lm;            // A-fragment row

  v8f accq[4] = {}; v8f acck[4] = {}; v8f accv[4] = {};

  for (int kc = 0; kc < D_ / 32; ++kc) {
    const int k0 = kc * 32;
    // A fragment: lane holds row arow; e -> K = half*16 + e
    const __bf16* xp = xb + (size_t)arow * D_ + k0;
    const v16bf a = cat16(*(const v8bf*)(xp + half * 16),
                          *(const v8bf*)(xp + half * 16 + 8));
    #pragma unroll
    for (int t = 0; t < 4; ++t) {
      // B fragment: contiguous 16 bf16 at packed offset (e -> K = half*16+e)
      const size_t fo = ((size_t)(kc * (D_ / 16) + g * 4 + t) * 16 + lm) * 32
                        + half * 16;
      const v16bf fbq = cat16(*(const v8bf*)(Wqp + fo), *(const v8bf*)(Wqp + fo + 8));
      const v16bf fbk = cat16(*(const v8bf*)(Wkp + fo), *(const v8bf*)(Wkp + fo + 8));
      const v16bf fbv = cat16(*(const v8bf*)(Wvp + fo), *(const v8bf*)(Wvp + fo + 8));
      accq[t] = wmma_bf16(a, fbq, accq[t]);
      acck[t] = wmma_bf16(a, fbk, acck[t]);
      accv[t] = wmma_bf16(a, fbv, accv[t]);
    }
  }

  // Epilogue: C/D layout is row = r + 8*half, col = lm (within 16x16 tile)
  const int b  = (mt * 16) / S_;            // tiles never straddle a batch
  const int s0 = (mt * 16) % S_;
  #pragma unroll
  for (int t = 0; t < 4; ++t) {
    const int hd  = t * 16 + lm;
    const float bqn = bq[g * 64 + hd];
    const float bvn = bv[g * 64 + hd];
    #pragma unroll
    for (int r = 0; r < 8; ++r) {
      const int s = s0 + r + 8 * half;
      const size_t qi = ((size_t)(b * H_ + g) * S_ + s) * HD_ + hd;
      qws[qi] = (__bf16)(accq[t][r] + bqn);
      kws[qi] = (__bf16)(acck[t][r]);
      const size_t vi = ((size_t)(b * H_ + g) * HD_ + hd) * S_ + s;
      vtws[vi] = (__bf16)(accv[t][r] + bvn);
    }
  }
}

// ---------------------------------------------------------------------------
// Kernel 2: flash-style causal attention.
// One wave owns a 16-row Q tile of one (b,h). Iterates k in chunks of 32:
//   S = scale * Q K^T  (2 WMMA over HD=64 per 16-col tile)
//   write S (+ causal -inf) tile to qk output exactly once
//   online softmax (row stats replicated per-lane via shfl_xor reductions)
//   P staged through LDS (C-layout -> A-layout transpose), O += P V
// Chunks entirely above the diagonal only write -inf (no compute).
// grid = (128/8, 48), block = 256.
// ---------------------------------------------------------------------------
__global__ __launch_bounds__(256) void attn_kernel(
    const __bf16* __restrict__ qws, const __bf16* __restrict__ kws,
    const __bf16* __restrict__ vtws,
    float* __restrict__ qk_out,            // [B,H,S,S] fp32 (masked logits)
    __bf16* __restrict__ ctx)              // [B*S, D], col = h*64+hd
{
  __shared__ __bf16 lds_p[8][16 * 32];     // per-wave P staging (1 KB each)
  const int lane = threadIdx.x & 31;
  const int wv   = threadIdx.x >> 5;
  const int half = lane >> 4;
  const int lm   = lane & 15;
  const int qt   = blockIdx.x * 8 + wv;    // q tile: 0..127
  const int bh   = blockIdx.y;             // 0..47
  const int b = bh / H_, h = bh % H_;
  __bf16* lp = lds_p[wv];

  const __bf16* qhead  = qws  + (size_t)bh * S_ * HD_;
  const __bf16* khead  = kws  + (size_t)bh * S_ * HD_;
  const __bf16* vthead = vtws + (size_t)bh * HD_ * S_;
  float* qkh = qk_out + ((size_t)bh << 22);          // bh * S * S

  const float scale = 0.125f;                        // 64^-0.5

  // Q A-fragments for both 32-wide d-chunks (held for the whole k loop)
  v16bf aqf[2];
  {
    const __bf16* qp = qhead + (size_t)(qt * 16 + lm) * HD_;
    #pragma unroll
    for (int c = 0; c < 2; ++c) {
      v8bf lo = *(const v8bf*)(qp + c * 32 + half * 16);
      v8bf hi = *(const v8bf*)(qp + c * 32 + half * 16 + 8);
      aqf[c] = cat16(lo, hi);
    }
  }

  v8f o[4] = {};
  float mrow[8], lrow[8];
  #pragma unroll
  for (int r = 0; r < 8; ++r) { mrow[r] = -__builtin_inff(); lrow[r] = 0.f; }

  const int qmax = qt * 16 + 15;
  for (int c = 0; c < S_ / 32; ++c) {
    const int kb = c * 32;
    if (kb > qmax) {                 // fully masked chunk: write -inf only
      #pragma unroll
      for (int t = 0; t < 2; ++t) {
        const int kcol = kb + t * 16 + lm;
        #pragma unroll
        for (int r = 0; r < 8; ++r) {
          const int qrow = qt * 16 + r + 8 * half;
          qkh[(size_t)qrow * S_ + kcol] = -__builtin_inff();
        }
      }
      continue;
    }

    // Prefetch next chunk's K rows and V columns while this chunk computes.
    if (kb + 32 <= qmax) {
      __builtin_prefetch(khead + (size_t)(kb + 32 + lm) * HD_, 0, 1);
      __builtin_prefetch(vthead + (size_t)lane * S_ + kb + 32, 0, 1);
    }

    // ---- S = Q K^T over HD=64 (two chained WMMAs per 16-col tile) ----
    v8f sc[2];
    #pragma unroll
    for (int t = 0; t < 2; ++t) {
      const __bf16* kp = khead + (size_t)(kb + t * 16 + lm) * HD_;  // lane = k row
      v8bf lo0 = *(const v8bf*)(kp + half * 16);
      v8bf hi0 = *(const v8bf*)(kp + half * 16 + 8);
      v8bf lo1 = *(const v8bf*)(kp + 32 + half * 16);
      v8bf hi1 = *(const v8bf*)(kp + 32 + half * 16 + 8);
      v8f z = {};
      z = wmma_bf16(aqf[0], cat16(lo0, hi0), z);
      z = wmma_bf16(aqf[1], cat16(lo1, hi1), z);
      sc[t] = z;
    }

    // ---- scale + causal mask + single write of qk tile ----
    #pragma unroll
    for (int t = 0; t < 2; ++t) {
      const int kcol = kb + t * 16 + lm;
      #pragma unroll
      for (int r = 0; r < 8; ++r) {
        const int qrow = qt * 16 + r + 8 * half;
        float v = sc[t][r] * scale;
        if (kcol > qrow) v = -__builtin_inff();
        sc[t][r] = v;
        qkh[(size_t)qrow * S_ + kcol] = v;
      }
    }

    // ---- online softmax: row max over 16 lanes of each half ----
    float alpha[8];
    #pragma unroll
    for (int r = 0; r < 8; ++r) {
      float v = fmaxf(sc[0][r], sc[1][r]);
      v = fmaxf(v, __shfl_xor(v, 1, 32));
      v = fmaxf(v, __shfl_xor(v, 2, 32));
      v = fmaxf(v, __shfl_xor(v, 4, 32));
      v = fmaxf(v, __shfl_xor(v, 8, 32));
      const float mn = fmaxf(mrow[r], v);   // finite: kb <= qt*16 => diag present
      alpha[r] = __expf(mrow[r] - mn);
      mrow[r]  = mn;
    }

    // ---- P = exp(S - m); stage bf16 P to LDS in row-major for A-layout read
    #pragma unroll
    for (int t = 0; t < 2; ++t) {
      #pragma unroll
      for (int r = 0; r < 8; ++r) {
        const float p = __expf(sc[t][r] - mrow[r]);   // exp(-inf)=0 for masked
        sc[t][r] = p;
        lp[(r + 8 * half) * 32 + t * 16 + lm] = (__bf16)p;
      }
    }
    #pragma unroll
    for (int r = 0; r < 8; ++r) {
      float v = sc[0][r] + sc[1][r];
      v += __shfl_xor(v, 1, 32);
      v += __shfl_xor(v, 2, 32);
      v += __shfl_xor(v, 4, 32);
      v += __shfl_xor(v, 8, 32);
      lrow[r] = lrow[r] * alpha[r] + v;
    }

    asm volatile("s_wait_dscnt 0" ::: "memory");      // P stores -> P loads
    v16bf pa;                                         // P as A-fragment (16x32)
    {
      v8bf lo = *(const v8bf*)(lp + lm * 32 + half * 16);
      v8bf hi = *(const v8bf*)(lp + lm * 32 + half * 16 + 8);
      pa = cat16(lo, hi);
    }

    // ---- rescale O by alpha, then O += P * V (K=32, N=64 via 4 tiles) ----
    #pragma unroll
    for (int t = 0; t < 4; ++t) {
      #pragma unroll
      for (int r = 0; r < 8; ++r) o[t][r] *= alpha[r];
    }
    #pragma unroll
    for (int t = 0; t < 4; ++t) {
      const __bf16* vp = vthead + (size_t)(t * 16 + lm) * S_ + kb; // lane = d col
      v8bf lo = *(const v8bf*)(vp + half * 16);
      v8bf hi = *(const v8bf*)(vp + half * 16 + 8);
      o[t] = wmma_bf16(pa, cat16(lo, hi), o[t]);
    }
  }

  // ---- normalize and write context (bf16, head-concatenated layout) ----
  #pragma unroll
  for (int t = 0; t < 4; ++t) {
    #pragma unroll
    for (int r = 0; r < 8; ++r) {
      const int srow = qt * 16 + r + 8 * half;
      const float v = o[t][r] / lrow[r];
      ctx[((size_t)(b * S_ + srow)) * D_ + h * HD_ + t * 16 + lm] = (__bf16)v;
    }
  }
}

// ---------------------------------------------------------------------------
// Kernel 3: output projection  out = ctx @ Wo + bo  (fp32 output to d_out).
// Wo pre-packed bf16. grid = (512/8, 12), block = 256.
// ---------------------------------------------------------------------------
__global__ __launch_bounds__(256) void out_proj_kernel(
    const __bf16* __restrict__ ctx,
    const __bf16* __restrict__ Wop, const float* __restrict__ bo,
    float* __restrict__ out)
{
  const int lane = threadIdx.x & 31;
  const int wv   = threadIdx.x >> 5;
  const int half = lane >> 4;
  const int lm   = lane & 15;
  const int mt   = blockIdx.x * 8 + wv;
  const int g    = blockIdx.y;
  const int arow = mt * 16 + lm;

  v8f acc[4] = {};
  for (int kc = 0; kc < D_ / 32; ++kc) {
    const int k0 = kc * 32;
    const __bf16* p = ctx + (size_t)arow * D_ + k0;
    const v16bf a = cat16(*(const v8bf*)(p + half * 16),
                          *(const v8bf*)(p + half * 16 + 8));
    #pragma unroll
    for (int t = 0; t < 4; ++t) {
      const size_t fo = ((size_t)(kc * (D_ / 16) + g * 4 + t) * 16 + lm) * 32
                        + half * 16;
      const v16bf bf = cat16(*(const v8bf*)(Wop + fo), *(const v8bf*)(Wop + fo + 8));
      acc[t] = wmma_bf16(a, bf, acc[t]);
    }
  }
  #pragma unroll
  for (int t = 0; t < 4; ++t) {
    const int n = g * 64 + t * 16 + lm;
    const float bias = bo[n];
    #pragma unroll
    for (int r = 0; r < 8; ++r) {
      const int row = mt * 16 + r + 8 * half;
      out[(size_t)row * D_ + n] = acc[t][r] + bias;
    }
  }
}

// ---------------------------------------------------------------------------
// Launch. d_out = [output (B*S*D) | qk (B*H*S*S)] fp32.
// Workspace (bf16 elements): qws|kws|vtws|ctx|xb (5 x B*S*D) then
// Wq|Wk|Wv|Wo packed (4 x D*D). Total ~67.6 MB.
// ---------------------------------------------------------------------------
extern "C" void kernel_launch(void* const* d_in, const int* in_sizes, int n_in,
                              void* d_out, int out_size, void* d_ws, size_t ws_size,
                              hipStream_t stream) {
  const float* x  = (const float*)d_in[0];
  const float* Wq = (const float*)d_in[1];
  const float* bq = (const float*)d_in[2];
  const float* Wk = (const float*)d_in[3];
  const float* Wv = (const float*)d_in[4];
  const float* bv = (const float*)d_in[5];
  const float* Wo = (const float*)d_in[6];
  const float* bo = (const float*)d_in[7];

  float* out = (float*)d_out;
  float* qk  = out + (size_t)B_ * S_ * D_;

  const size_t n_bsd = (size_t)B_ * S_ * D_;   // 6,291,456
  const size_t n_w   = (size_t)D_ * D_;        // 589,824
  __bf16* qws  = (__bf16*)d_ws;
  __bf16* kws  = qws  + n_bsd;
  __bf16* vtws = kws  + n_bsd;
  __bf16* ctx  = vtws + n_bsd;
  __bf16* xb   = ctx  + n_bsd;
  __bf16* wqp  = xb   + n_bsd;
  __bf16* wkp  = wqp  + n_w;
  __bf16* wvp  = wkp  + n_w;
  __bf16* wop  = wvp  + n_w;

  pack_x_kernel<<<dim3((unsigned)(n_bsd / (256 * 4))), 256, 0, stream>>>(x, xb);
  pack_w_kernel<<<dim3((unsigned)(n_w / 256)), 256, 0, stream>>>(Wq, wqp);
  pack_w_kernel<<<dim3((unsigned)(n_w / 256)), 256, 0, stream>>>(Wk, wkp);
  pack_w_kernel<<<dim3((unsigned)(n_w / 256)), 256, 0, stream>>>(Wv, wvp);
  pack_w_kernel<<<dim3((unsigned)(n_w / 256)), 256, 0, stream>>>(Wo, wop);

  proj_qkv_kernel<<<dim3(64, 12), 256, 0, stream>>>(xb, wqp, bq, wkp, wvp, bv,
                                                    qws, kws, vtws);
  attn_kernel<<<dim3(16, 48), 256, 0, stream>>>(qws, kws, vtws, qk, ctx);
  out_proj_kernel<<<dim3(64, 12), 256, 0, stream>>>(ctx, wop, bo, out);
}
